// GGNNEncoder_1271310320378
// MI455X (gfx1250) — matrix operand order repaired and checked by reading, not compiled, not run on hardware
//
#include <hip/hip_runtime.h>
#include <hip/hip_bf16.h>
#include <stdint.h>

// ---------------------------------------------------------------------------
// Types
// ---------------------------------------------------------------------------
typedef __bf16 bf16_t;
typedef __attribute__((ext_vector_type(16))) __bf16 v16bf;
typedef __attribute__((ext_vector_type(8)))  float  v8f;
typedef __attribute__((ext_vector_type(4)))  unsigned int uint4v;

#define TMB 256  // workgroup tile M
#define TNB 128  // workgroup tile N
#define TK  32   // K chunk
#define LP  40   // LDS pitch (bf16 elems); %8==0 keeps 16B alignment for b128 ops

__device__ __forceinline__ unsigned short f32_to_bf16_bits(float f) {
    union { float f; unsigned int u; } v; v.f = f;
    unsigned int u = v.u;
    u += 0x7FFFu + ((u >> 16) & 1u);          // round-to-nearest-even
    return (unsigned short)(u >> 16);
}

// ---------------------------------------------------------------------------
// Generic batched bf16 GEMM, f32 accumulate via v_wmma_f32_16x16x32_bf16.
//   C[z] = act( A[z/aDiv] (MxK) @ B[z%bMod] (KxN) + bias[z%biasMod] )
// Requirements (all satisfied here): M%256==0, N%128==0, K%32==0.
// 256 threads = 8 waves, each wave owns a 64x64 C tile (4x4 WMMA accs).
// Double-buffered LDS staging: global prefetch of chunk k+1 overlaps the
// matrix pipe on chunk k; one workgroup barrier per K chunk.
// ---------------------------------------------------------------------------
__global__ __launch_bounds__(256)
void gemm_bf16_wmma(const bf16_t* __restrict__ A, const bf16_t* __restrict__ B,
                    const float* __restrict__ bias, void* __restrict__ C,
                    int M, int N, int K,
                    long aStride, int aDiv, long bStride, int bMod,
                    long cStride, int ldc, int biasMod, int act, int outBf16)
{
    __shared__ bf16_t ldsA[2][TMB * LP];   // 2 x 20 KB
    __shared__ bf16_t ldsB[2][TNB * LP];   // 2 x 10 KB (stored transposed [n][k])

    const int t     = threadIdx.x;
    const int bz    = blockIdx.z;
    const int tileM = blockIdx.y * TMB;
    const int tileN = blockIdx.x * TNB;

    const bf16_t* Ab = A + (long)(bz / aDiv) * aStride;
    const bf16_t* Bb = B + (long)(bz % bMod) * bStride;

    const int wid  = t >> 5;        // 0..7
    const int lane = t & 31;
    const int wm   = wid & 3;       // 4 waves along M -> 64 rows each
    const int wn   = wid >> 2;      // 2 waves along N -> 64 cols each
    const int lrow = lane & 15;
    const int lhi  = lane >> 4;     // half-wave select

    // per-thread staging coordinates (constant across chunks)
    int aRow[4], aCol[4];
#pragma unroll
    for (int i = 0; i < 4; ++i) {
        int e0 = (t + i * 256) << 3;       // 8 bf16 per b128
        aRow[i] = e0 >> 5;                 // /TK
        aCol[i] = e0 & 31;
    }
    int bKr[2], bNc[2];
#pragma unroll
    for (int i = 0; i < 2; ++i) {
        int e0 = (t + i * 256) << 3;
        bKr[i] = e0 >> 7;                  // /TNB
        bNc[i] = e0 & 127;
    }

    v8f acc[4][4];
#pragma unroll
    for (int i = 0; i < 4; ++i)
#pragma unroll
        for (int j = 0; j < 4; ++j)
#pragma unroll
            for (int v = 0; v < 8; ++v) acc[i][j][v] = 0.0f;

    const int nk = K / TK;
    uint4v ra[4], rb[2];

    // ---- prologue: fetch + stage chunk 0 ----
#pragma unroll
    for (int i = 0; i < 4; ++i)
        ra[i] = *(const uint4v*)(Ab + (long)(tileM + aRow[i]) * K + aCol[i]);
#pragma unroll
    for (int i = 0; i < 2; ++i)
        rb[i] = *(const uint4v*)(Bb + (long)bKr[i] * N + (tileN + bNc[i]));
#pragma unroll
    for (int i = 0; i < 4; ++i)
        *(uint4v*)&ldsA[0][aRow[i] * LP + aCol[i]] = ra[i];
#pragma unroll
    for (int i = 0; i < 2; ++i) {
        bf16_t tmp[8]; *(uint4v*)tmp = rb[i];
#pragma unroll
        for (int j = 0; j < 8; ++j)
            ldsB[0][(bNc[i] + j) * LP + bKr[i]] = tmp[j];
    }
    __syncthreads();

    for (int kc = 0; kc < nk; ++kc) {
        const int cur  = kc & 1;
        const int nxt  = cur ^ 1;
        const bool more = (kc + 1) < nk;
        const int k1 = (kc + 1) * TK;

        // ---- prefetch chunk k+1 from global into registers ----
        if (more) {
#pragma unroll
            for (int i = 0; i < 4; ++i)
                ra[i] = *(const uint4v*)(Ab + (long)(tileM + aRow[i]) * K + (k1 + aCol[i]));
#pragma unroll
            for (int i = 0; i < 2; ++i)
                rb[i] = *(const uint4v*)(Bb + (long)(k1 + bKr[i]) * N + (tileN + bNc[i]));
        }

        // ---- fragments per documented 16-bit WMMA VGPR layouts ----
        v16bf afrag[4], bfrag[4];
#pragma unroll
        for (int i = 0; i < 4; ++i) {
            int m  = wm * 64 + i * 16 + lrow;
            int kg = lhi * 8;
            union { uint4v u[2]; v16bf v; } U;
            U.u[0] = *(const uint4v*)&ldsA[cur][m * LP + kg];        // K kg..kg+7   -> V0..V3
            U.u[1] = *(const uint4v*)&ldsA[cur][m * LP + 16 + kg];   // K 16+kg..+7  -> V4..V7
            afrag[i] = U.v;
        }
#pragma unroll
        for (int j = 0; j < 4; ++j) {
            int n  = wn * 64 + j * 16 + lrow;
            int kb = lhi * 8;
            union { uint4v u[2]; v16bf v; } U;
            U.u[0] = *(const uint4v*)&ldsB[cur][n * LP + kb];        // K kb..kb+7
            U.u[1] = *(const uint4v*)&ldsB[cur][n * LP + kb + 16];   // K 16+kb..+7
            bfrag[j] = U.v;
        }

        // ---- 16 WMMAs on the matrix pipe ----
#pragma unroll
        for (int i = 0; i < 4; ++i)
#pragma unroll
            for (int j = 0; j < 4; ++j)
                acc[i][j] = __builtin_amdgcn_wmma_f32_16x16x32_bf16(
                    false, afrag[i], false, bfrag[j], (short)0, acc[i][j], false, false);

        // ---- stage chunk k+1 into the other LDS buffer ----
        if (more) {
#pragma unroll
            for (int i = 0; i < 4; ++i)
                *(uint4v*)&ldsA[nxt][aRow[i] * LP + aCol[i]] = ra[i];
#pragma unroll
            for (int i = 0; i < 2; ++i) {
                bf16_t tmp[8]; *(uint4v*)tmp = rb[i];
#pragma unroll
                for (int j = 0; j < 8; ++j)
                    ldsB[nxt][(bNc[i] + j) * LP + bKr[i]] = tmp[j];
            }
        }
        __syncthreads();
    }

    // ---- epilogue: bias + activation, f32 or bf16 out, strided ldc ----
    const float* bb = bias ? bias + (long)(bz % biasMod) * N : nullptr;
    const long cBase = (long)bz * cStride;
#pragma unroll
    for (int i = 0; i < 4; ++i) {
#pragma unroll
        for (int j = 0; j < 4; ++j) {
            int n  = tileN + wn * 64 + j * 16 + lrow;
            float bv = bb ? bb[n] : 0.0f;
            int mBase = tileM + wm * 64 + i * 16 + lhi * 8;
#pragma unroll
            for (int v = 0; v < 8; ++v) {
                float x = acc[i][j][v] + bv;
                if      (act == 1) x = 1.0f / (1.0f + __expf(-x));      // sigmoid
                else if (act == 2) x = x > 0.0f ? x : 0.01f * x;        // leaky relu
                else if (act == 3) x = x > 0.0f ? x : 0.0f;             // relu
                long off = cBase + (long)(mBase + v) * ldc + n;
                if (outBf16) ((unsigned short*)C)[off] = f32_to_bf16_bits(x);
                else         ((float*)C)[off] = x;
            }
        }
    }
}

// ---------------------------------------------------------------------------
// Elementwise helpers
// ---------------------------------------------------------------------------
__global__ void convert_f32_bf16(const float* __restrict__ s,
                                 unsigned short* __restrict__ d, long n) {
    long i = (long)blockIdx.x * 256 + threadIdx.x;
    if (i < n) d[i] = f32_to_bf16_bits(s[i]);
}

// edges (B,N,2048) -> ein (B,N,1024) bf16, eout (B,N,1024) bf16
__global__ void edges_split(const float* __restrict__ e,
                            unsigned short* __restrict__ ein,
                            unsigned short* __restrict__ eout, long n) {
    long i = (long)blockIdx.x * 256 + threadIdx.x;
    if (i >= n) return;
    long bn = i >> 10; int c = (int)(i & 1023);
    const float* row = e + bn * 2048;
    ein[i]  = f32_to_bf16_bits(row[c]);
    eout[i] = f32_to_bf16_bits(row[1024 + c]);
}

// state f32 -> state bf16, and abuf[...,1024+d] = state (concat tail)
__global__ void state_prep(const float* __restrict__ st,
                           unsigned short* __restrict__ sbf,
                           unsigned short* __restrict__ abuf, long n) {
    long i = (long)blockIdx.x * 256 + threadIdx.x;
    if (i >= n) return;
    unsigned short v = f32_to_bf16_bits(st[i]);
    sbf[i] = v;
    long bn = i >> 9; int d = (int)(i & 511);
    abuf[bn * 1536 + 1024 + d] = v;
}

// abuf[...,1024+d] = bf16(r * state)   (the "joined" tail)
__global__ void joined_tail(const float* __restrict__ r, const float* __restrict__ st,
                            unsigned short* __restrict__ abuf, long n) {
    long i = (long)blockIdx.x * 256 + threadIdx.x;
    if (i >= n) return;
    unsigned short v = f32_to_bf16_bits(r[i] * st[i]);
    long bn = i >> 9; int d = (int)(i & 511);
    abuf[bn * 1536 + 1024 + d] = v;
}

// state = (1-z)*state + z*hhat   (in place)
__global__ void gru_update(float* __restrict__ st, const float* __restrict__ z,
                           const float* __restrict__ h, long n) {
    long i = (long)blockIdx.x * 256 + threadIdx.x;
    if (i >= n) return;
    float zi = z[i];
    st[i] = (1.0f - zi) * st[i] + zi * h[i];
}

// join_bf16[n,d] = bf16(mean_b state[b,n,d])
__global__ void batch_mean(const float* __restrict__ st,
                           unsigned short* __restrict__ jb, long nd) {
    long i = (long)blockIdx.x * 256 + threadIdx.x;
    if (i >= nd) return;
    float s = 0.0f;
    for (int b = 0; b < 32; ++b) s += st[(long)b * nd + i];
    jb[i] = f32_to_bf16_bits(s * (1.0f / 32.0f));
}

// ---------------------------------------------------------------------------
// Host orchestration
// ---------------------------------------------------------------------------
static void launch_gemm(hipStream_t s, const void* A, const void* B, const float* bias,
                        void* C, int M, int N, int K, int batch,
                        long aS, int aDiv, long bS, int bMod,
                        long cS, int ldc, int biasMod, int act, int outBf16) {
    dim3 grid(N / TNB, M / TMB, batch);
    gemm_bf16_wmma<<<grid, 256, 0, s>>>((const bf16_t*)A, (const bf16_t*)B, bias, C,
                                        M, N, K, aS, aDiv, bS, bMod, cS, ldc,
                                        biasMod, act, outBf16);
}

extern "C" void kernel_launch(void* const* d_in, const int* in_sizes, int n_in,
                              void* d_out, int out_size, void* d_ws, size_t ws_size,
                              hipStream_t stream) {
    (void)in_sizes; (void)n_in; (void)out_size; (void)ws_size;
    constexpr long Bsz = 32, Nn = 256, Dd = 512, Ee = 4;
    constexpr long S    = Bsz * Nn * Dd;          // 4,194,304 (state elems)
    constexpr long EN   = Ee * Nn;                // 1024
    constexpr long EDG  = Bsz * Nn * EN;          // 8,388,608
    constexpr long INS  = Bsz * EN * Dd;          // 16,777,216
    constexpr long ABUF = Bsz * Nn * 3 * Dd;      // 12,582,912
    constexpr long ND   = Nn * Dd;                // 131,072

    const float* prop   = (const float*)d_in[0];
    const float* edges  = (const float*)d_in[1];
    const float* W_in   = (const float*)d_in[2];
    const float* b_in   = (const float*)d_in[3];
    const float* W_out  = (const float*)d_in[4];
    const float* b_out  = (const float*)d_in[5];
    const float* W_r    = (const float*)d_in[6];
    const float* b_r    = (const float*)d_in[7];
    const float* W_z    = (const float*)d_in[8];
    const float* b_z    = (const float*)d_in[9];
    const float* W_h    = (const float*)d_in[10];
    const float* b_h    = (const float*)d_in[11];
    const float* W_bh   = (const float*)d_in[12];
    const float* b_bh   = (const float*)d_in[13];
    const float* W_bc   = (const float*)d_in[14];
    const float* b_bc   = (const float*)d_in[15];

    // workspace carve-up (bf16 buffers as ushort)
    uint8_t* w = (uint8_t*)d_ws;
    auto carve = [&](long bytes) { void* p = w; w += (bytes + 255) & ~255L; return p; };
    unsigned short* stateBf = (unsigned short*)carve(S * 2);
    unsigned short* einBf   = (unsigned short*)carve(EDG * 2);
    unsigned short* eoutBf  = (unsigned short*)carve(EDG * 2);
    unsigned short* WinBf   = (unsigned short*)carve(Ee * Dd * Dd * 2);
    unsigned short* WoutBf  = (unsigned short*)carve(Ee * Dd * Dd * 2);
    unsigned short* WrBf    = (unsigned short*)carve(3 * Dd * Dd * 2);
    unsigned short* WzBf    = (unsigned short*)carve(3 * Dd * Dd * 2);
    unsigned short* WhBf    = (unsigned short*)carve(3 * Dd * Dd * 2);
    unsigned short* WbhBf   = (unsigned short*)carve(Dd * Dd * 2);
    unsigned short* WbcBf   = (unsigned short*)carve(Dd * Dd * 2);
    unsigned short* insBf   = (unsigned short*)carve(INS * 2);
    unsigned short* outsBf  = (unsigned short*)carve(INS * 2);
    unsigned short* abuf    = (unsigned short*)carve(ABUF * 2);
    float*          rbuf    = (float*)carve(S * 4);
    float*          zbuf    = (float*)carve(S * 4);
    float*          hbuf    = (float*)carve(S * 4);
    unsigned short* joinBf  = (unsigned short*)carve(ND * 2);

    float* state = (float*)d_out;                 // live state in output slot
    float* outH  = (float*)d_out + S;
    float* outC  = (float*)d_out + S + 4 * ND;

    auto blocks = [](long n) { return (unsigned)((n + 255) / 256); };

    // init state from input; convert edges + weights to bf16 (once per call)
    hipMemcpyAsync(state, prop, S * sizeof(float), hipMemcpyDeviceToDevice, stream);
    edges_split<<<blocks(EDG), 256, 0, stream>>>(edges, einBf, eoutBf, EDG);
    convert_f32_bf16<<<blocks(Ee*Dd*Dd), 256, 0, stream>>>(W_in,  WinBf,  Ee*Dd*Dd);
    convert_f32_bf16<<<blocks(Ee*Dd*Dd), 256, 0, stream>>>(W_out, WoutBf, Ee*Dd*Dd);
    convert_f32_bf16<<<blocks(3*Dd*Dd), 256, 0, stream>>>(W_r, WrBf, 3*Dd*Dd);
    convert_f32_bf16<<<blocks(3*Dd*Dd), 256, 0, stream>>>(W_z, WzBf, 3*Dd*Dd);
    convert_f32_bf16<<<blocks(3*Dd*Dd), 256, 0, stream>>>(W_h, WhBf, 3*Dd*Dd);
    convert_f32_bf16<<<blocks(Dd*Dd), 256, 0, stream>>>(W_bh, WbhBf, Dd*Dd);
    convert_f32_bf16<<<blocks(Dd*Dd), 256, 0, stream>>>(W_bc, WbcBf, Dd*Dd);

    for (int step = 0; step < 4; ++step) {
        // state -> bf16 (+ concat tail = state)
        state_prep<<<blocks(S), 256, 0, stream>>>(state, stateBf, abuf, S);
        // ins[b,e] = state[b] @ W_in[e] + b_in[e]   (batch = b*4+e)
        launch_gemm(stream, stateBf, WinBf, b_in, insBf, 256, 512, 512, 128,
                    ND, 4, Dd*Dd, 4, ND, 512, 4, 0, 1);
        launch_gemm(stream, stateBf, WoutBf, b_out, outsBf, 256, 512, 512, 128,
                    ND, 4, Dd*Dd, 4, ND, 512, 4, 0, 1);
        // a_in[b] = edges_in[b] @ ins[b] -> abuf cols [0,512)
        launch_gemm(stream, einBf, insBf, nullptr, abuf, 256, 512, 1024, 32,
                    Nn*EN, 1, EN*Dd, 32, (long)Nn*3*Dd, 1536, 1, 0, 1);
        // a_out[b] = edges_out[b] @ outs[b] -> abuf cols [512,1024)
        launch_gemm(stream, eoutBf, outsBf, nullptr, abuf + 512, 256, 512, 1024, 32,
                    Nn*EN, 1, EN*Dd, 32, (long)Nn*3*Dd, 1536, 1, 0, 1);
        // r = sigmoid(a @ W_r + b_r) ; z = sigmoid(a @ W_z + b_z)
        launch_gemm(stream, abuf, WrBf, b_r, rbuf, 256, 512, 1536, 32,
                    (long)Nn*3*Dd, 1, 0, 1, ND, 512, 1, 1, 0);
        launch_gemm(stream, abuf, WzBf, b_z, zbuf, 256, 512, 1536, 32,
                    (long)Nn*3*Dd, 1, 0, 1, ND, 512, 1, 1, 0);
        // joined tail = r * state
        joined_tail<<<blocks(S), 256, 0, stream>>>(rbuf, state, abuf, S);
        // h_hat = leaky_relu(joined @ W_h + b_h)
        launch_gemm(stream, abuf, WhBf, b_h, hbuf, 256, 512, 1536, 32,
                    (long)Nn*3*Dd, 1, 0, 1, ND, 512, 1, 2, 0);
        // state = (1-z)*state + z*h_hat
        gru_update<<<blocks(S), 256, 0, stream>>>(state, zbuf, hbuf, S);
    }

    // join = mean over batch; bridge heads (broadcast 4x via batch w/ aStride=0)
    batch_mean<<<blocks(ND), 256, 0, stream>>>(state, joinBf, ND);
    launch_gemm(stream, joinBf, WbhBf, b_bh, outH, 256, 512, 512, 4,
                0, 1, 0, 1, ND, 512, 1, 3, 0);
    launch_gemm(stream, joinBf, WbcBf, b_bc, outC, 256, 512, 512, 4,
                0, 1, 0, 1, ND, 512, 1, 3, 0);
}
